// ProjResNet_53807350284655
// MI455X (gfx1250) — matrix operand order repaired and checked by reading, not compiled, not run on hardware
//
#include <hip/hip_runtime.h>

// ---------------- problem constants ----------------
constexpr int Bn   = 1024;          // samples
constexpr int Dn   = 1024;          // feature dim
constexpr int KR   = 16;            // low rank
constexpr int DIMn = 64;            // Dn / KR
constexpr int Ln   = 16;            // residual layers
constexpr int NC   = 10;            // classes
constexpr int ROWS = Bn * 3;        // 3072 channel-rows

// ---------------- vector types ----------------
typedef __attribute__((ext_vector_type(16))) __bf16        v16bf;
typedef __attribute__((ext_vector_type(8)))  float         v8f;
typedef __attribute__((ext_vector_type(2)))  float         v2f;
typedef __attribute__((ext_vector_type(4)))  unsigned int  v4u;
typedef __attribute__((ext_vector_type(8)))  int           v8i;
typedef __attribute__((ext_vector_type(4)))  int           v4i;

union BFrag { uint4 q[2]; v16bf v; };

__device__ __forceinline__ unsigned short f2bf(float f) {
    unsigned int u = __float_as_uint(f);
    unsigned int r = u + 0x7FFFu + ((u >> 16) & 1u);   // round-to-nearest-even
    return (unsigned short)(r >> 16);
}

// ---------------------------------------------------------------------
// TDM: async 2D tile load (32 x `rows` bf16 elements, tensor row stride
// 1024 elements) from global to LDS.  D# bitfields per CDNA5 ISA §8.3-8.6.
// ---------------------------------------------------------------------
__device__ __forceinline__ void tdm_load_tile(unsigned lds_off, const void* gptr, int rows) {
    unsigned long long ga = (unsigned long long)(uintptr_t)gptr;
    v4u g0 = { 1u,                                   // count = 1 valid descriptor
               lds_off,                              // LDS byte address
               (unsigned)ga,                         // global addr lo
               (unsigned)((ga >> 32) & 0x01FFFFFFu) | (2u << 30) };  // hi | type=2
    v8i g1 = { (int)(1u << 16),                      // wg_mask=0, data_size=1 (2B)
               (int)(1024u << 16),                   // abar=0 | tensor_dim0[15:0]
               (int)(rows << 16),                    // td0[31:16]=0 | tensor_dim1[15:0]
               (int)(32u << 16),                     // td1[31:16]=0 | tile_dim0=32
               (int)rows,                            // tile_dim1=rows | tile_dim2=0
               (int)1024,                            // tensor_dim0_stride[31:0]
               0, 0 };
    v4i z4 = {0, 0, 0, 0};
#if __clang_major__ >= 23
    v8i z8 = {0, 0, 0, 0, 0, 0, 0, 0};
    __builtin_amdgcn_tensor_load_to_lds(g0, g1, z4, z4, z8, 0);
#else
    __builtin_amdgcn_tensor_load_to_lds(g0, g1, z4, z4, 0);
#endif
}

__device__ __forceinline__ unsigned lds_offset_of(const void* p) {
    return (unsigned)(uintptr_t)p;                   // low 32 bits = LDS byte offset
}

// =====================================================================
// prep: build Xcur (fp32) and Xbf (bf16) with channel-2 transpose
// =====================================================================
__global__ __launch_bounds__(256)
void prep_kernel(const float* __restrict__ Xin, float* __restrict__ Xcur,
                 unsigned short* __restrict__ Xbf) {
    int idx = blockIdx.x * 256 + threadIdx.x;           // over ROWS*Dn
    if (idx >= ROWS * Dn) return;
    int d  = idx & (Dn - 1);
    int bc = idx >> 10;
    int c  = bc % 3;
    float v;
    if (c == 2) {
        int b = bc / 3;
        v = Xin[(size_t)(b * 3 + 2) * Dn + (d & 15) * DIMn + (d >> 4)];
    } else {
        v = Xin[idx];
    }
    Xcur[idx] = v;
    Xbf[idx]  = f2bf(v);
}

// =====================================================================
// wconv: fp32 -> bf16 for W0 and all Ws
// =====================================================================
__global__ __launch_bounds__(256)
void wconv_kernel(const float* __restrict__ W0, const float* __restrict__ Ws,
                  unsigned short* __restrict__ Wbf) {
    int idx = blockIdx.x * 256 + threadIdx.x;           // over 17*1M
    if (idx >= (Ln + 1) * Dn * Dn) return;
    float v = (idx < Dn * Dn) ? W0[idx] : Ws[idx - Dn * Dn];
    Wbf[idx] = f2bf(v);
}

// =====================================================================
// gemm: Y[r,e] = sum_d A[r,d] * W[e,d]   (A bf16 ROWSxD, W bf16 DxD)
// Block = 4 waves; each wave computes 32 rows x 64 cols (2 A-frags,
// 8 accumulators) -> block covers 128 rows x 64 cols.  A (128x32) and
// W (64x32) K-slices are staged into LDS by the Tensor Data Mover
// (wave 0), double-buffered so the next-slice DMA overlaps WMMA.
// fused epilogue (fuse=1): Y = Xres + h * relu(acc + bias)
// grid = (Dn/64, ROWS/128)
// =====================================================================
__global__ __launch_bounds__(128)
void gemm_kernel(const unsigned short* __restrict__ A,
                 const unsigned short* __restrict__ W,
                 const float* __restrict__ bias,
                 const float* __restrict__ Xres,
                 const float* __restrict__ hptr,
                 float* __restrict__ Yout,
                 int fuse) {
    __shared__ unsigned short sA[2][128 * 32];          // 8 KB x 2
    __shared__ unsigned short sW[2][64 * 32];           // 4 KB x 2

    const int lane = threadIdx.x & 31;
    const int wave = threadIdx.x >> 5;
    const int rowBase = blockIdx.y * 128;
    const int colBase = blockIdx.x * 64;
    const int nn = lane & 15;
    const int hi = lane >> 4;

    const unsigned short* Ag = A + (size_t)rowBase * Dn;  // 128 rows
    const unsigned short* Wg = W + (size_t)colBase * Dn;  // 64 "cols" (rows of W)

    v8f acc[2][4];
    const v8f vz = {0.f,0.f,0.f,0.f,0.f,0.f,0.f,0.f};
#pragma unroll
    for (int s = 0; s < 2; ++s)
#pragma unroll
        for (int ct = 0; ct < 4; ++ct) acc[s][ct] = vz;

    // prologue: DMA first K-slice
    if (wave == 0) {
        tdm_load_tile(lds_offset_of(&sA[0][0]), Ag, 128);
        tdm_load_tile(lds_offset_of(&sW[0][0]), Wg, 64);
        __builtin_amdgcn_s_wait_tensorcnt(0);
    }
    __syncthreads();

    for (int k = 0; k < Dn; k += 32) {
        const int cur = (k >> 5) & 1;
        const int nxt = cur ^ 1;
        if (k + 32 < Dn && wave == 0) {                 // prefetch next slice
            tdm_load_tile(lds_offset_of(&sA[nxt][0]), Ag + k + 32, 128);
            tdm_load_tile(lds_offset_of(&sW[nxt][0]), Wg + k + 32, 64);
        }

        // A fragments (16-bit A 16x32 layout): local rows wave*32 + {0,16} + (lane&15)
        BFrag a0, a1;
        {
            const int ar0 = wave * 32 + nn;
            const int ar1 = ar0 + 16;
            a0.q[0] = *(const uint4*)&sA[cur][ar0 * 32 + hi * 8];
            a0.q[1] = *(const uint4*)&sA[cur][ar0 * 32 + 16 + hi * 8];
            a1.q[0] = *(const uint4*)&sA[cur][ar1 * 32 + hi * 8];
            a1.q[1] = *(const uint4*)&sA[cur][ar1 * 32 + 16 + hi * 8];
        }
        // B fragments (32x16): same k order as A (k = hi*8.., 16 + hi*8..)
        BFrag b0, b1, b2, b3;
        b0.q[0] = *(const uint4*)&sW[cur][(0 * 16 + nn) * 32 + hi * 8];
        b0.q[1] = *(const uint4*)&sW[cur][(0 * 16 + nn) * 32 + 16 + hi * 8];
        b1.q[0] = *(const uint4*)&sW[cur][(1 * 16 + nn) * 32 + hi * 8];
        b1.q[1] = *(const uint4*)&sW[cur][(1 * 16 + nn) * 32 + 16 + hi * 8];
        b2.q[0] = *(const uint4*)&sW[cur][(2 * 16 + nn) * 32 + hi * 8];
        b2.q[1] = *(const uint4*)&sW[cur][(2 * 16 + nn) * 32 + 16 + hi * 8];
        b3.q[0] = *(const uint4*)&sW[cur][(3 * 16 + nn) * 32 + hi * 8];
        b3.q[1] = *(const uint4*)&sW[cur][(3 * 16 + nn) * 32 + 16 + hi * 8];

        // keep all fragment loads above the WMMA block -> single dscnt wait,
        // then 8 back-to-back v_wmma
        __builtin_amdgcn_sched_barrier(0);

        acc[0][0] = __builtin_amdgcn_wmma_f32_16x16x32_bf16(false, a0.v, false, b0.v, (short)0, acc[0][0], false, false);
        acc[0][1] = __builtin_amdgcn_wmma_f32_16x16x32_bf16(false, a0.v, false, b1.v, (short)0, acc[0][1], false, false);
        acc[0][2] = __builtin_amdgcn_wmma_f32_16x16x32_bf16(false, a0.v, false, b2.v, (short)0, acc[0][2], false, false);
        acc[0][3] = __builtin_amdgcn_wmma_f32_16x16x32_bf16(false, a0.v, false, b3.v, (short)0, acc[0][3], false, false);
        acc[1][0] = __builtin_amdgcn_wmma_f32_16x16x32_bf16(false, a1.v, false, b0.v, (short)0, acc[1][0], false, false);
        acc[1][1] = __builtin_amdgcn_wmma_f32_16x16x32_bf16(false, a1.v, false, b1.v, (short)0, acc[1][1], false, false);
        acc[1][2] = __builtin_amdgcn_wmma_f32_16x16x32_bf16(false, a1.v, false, b2.v, (short)0, acc[1][2], false, false);
        acc[1][3] = __builtin_amdgcn_wmma_f32_16x16x32_bf16(false, a1.v, false, b3.v, (short)0, acc[1][3], false, false);

        if (k + 32 < Dn) {
            if (wave == 0) __builtin_amdgcn_s_wait_tensorcnt(0);
            __syncthreads();
        }
    }

    const float hv = hptr[0];
#pragma unroll
    for (int s = 0; s < 2; ++s) {
#pragma unroll
        for (int ct = 0; ct < 4; ++ct) {
            const int col = colBase + ct * 16 + nn;
            const float bval = fuse ? bias[col] : 0.0f;
#pragma unroll
            for (int vg = 0; vg < 8; ++vg) {
                const int r = rowBase + wave * 32 + s * 16 + vg + hi * 8;
                float v = acc[s][ct][vg];
                if (fuse) {
                    v += bval;
                    v = v > 0.0f ? v : 0.0f;
                    v = Xres[(size_t)r * Dn + col] + hv * v;
                }
                Yout[(size_t)r * Dn + col] = v;
            }
        }
    }
}

// =====================================================================
// polar: Newton-Schulz polar projection of 64x16 matrices (channels 0,2);
// channel 1 pass-through.  One wave per channel-row; 8 waves / block.
// =====================================================================
#define PSTR 17
__global__ __launch_bounds__(256)
void polar_kernel(const float* __restrict__ Yin,
                  float* __restrict__ Xcur,
                  unsigned short* __restrict__ Xbf,
                  float* __restrict__ Xt,
                  int layer) {
    __shared__ float sA[8][DIMn * PSTR];
    __shared__ float sT[8][16 * PSTR];

    const int lane = threadIdx.x & 31;
    const int wave = threadIdx.x >> 5;
    const int ridx = blockIdx.x * 8 + wave;             // 0..3071
    const int c = ridx % 3;

    const float* src = Yin + (size_t)ridx * Dn;
    float* xc = Xcur + (size_t)ridx * Dn;
    unsigned short* xb = Xbf + (size_t)ridx * Dn;
    float* xt = Xt + (size_t)ridx * Dn * (Ln + 1) + layer;

    if (c == 1) {                                       // S channel: copy
        for (int i = 0; i < 32; ++i) {
            int d = i * 32 + lane;
            float v = src[d];
            xc[d] = v; xb[d] = f2bf(v);
            __builtin_nontemporal_store(v, &xt[(size_t)d * (Ln + 1)]);
        }
        return;
    }

    float* Am = sA[wave];
    float* Tm = sT[wave];

    float ss = 0.0f;
    for (int i = 0; i < 32; ++i) {
        int d = i * 32 + lane;
        float v = src[d];
        ss += v * v;
        Am[(d >> 4) * PSTR + (d & 15)] = v;
    }
    for (int off = 16; off; off >>= 1) ss += __shfl_xor(ss, off, 32);
    const float inv = (ss > 1e-30f) ? rsqrtf(ss) : 0.0f;
    for (int i = 0; i < 32; ++i) {
        int d = i * 32 + lane;
        Am[(d >> 4) * PSTR + (d & 15)] *= inv;
    }
    __builtin_amdgcn_wave_barrier();

    const int n  = lane & 15;
    const int hi = lane >> 4;
    const v8f vz = {0.f,0.f,0.f,0.f,0.f,0.f,0.f,0.f};

    for (int it = 0; it < 12; ++it) {
        // ---- G = A^T A via 16 x V_WMMA_F32_16X16X4_F32 ----
        v8f G = vz;
#pragma unroll
        for (int cc = 0; cc < 16; ++cc) {
            v2f f;
            const int r0 = 4 * cc + 2 * hi;
            f.x = Am[r0 * PSTR + n];
            f.y = Am[(r0 + 1) * PSTR + n];
            G = __builtin_amdgcn_wmma_f32_16x16x4_f32(
                false, f, false, f, (short)0, G, false, false);
        }
        // ---- T = 1.5 I - 0.5 G ----
#pragma unroll
        for (int vg = 0; vg < 8; ++vg) {
            const int mm = vg + 8 * hi;
            Tm[mm * PSTR + n] = ((mm == n) ? 1.5f : 0.0f) - 0.5f * G[vg];
        }
        __builtin_amdgcn_wave_barrier();

        // ---- A <- A @ T ----
        v2f bt[4];
#pragma unroll
        for (int kc = 0; kc < 4; ++kc) {
            const int r0 = 4 * kc + 2 * hi;
            bt[kc].x = Tm[r0 * PSTR + n];
            bt[kc].y = Tm[(r0 + 1) * PSTR + n];
        }
        v8f Cacc[4];
#pragma unroll
        for (int t = 0; t < 4; ++t) {
            v8f acc = vz;
#pragma unroll
            for (int kc = 0; kc < 4; ++kc) {
                v2f af;
                const int rr = 16 * t + n;
                af.x = Am[rr * PSTR + 4 * kc + 2 * hi];
                af.y = Am[rr * PSTR + 4 * kc + 2 * hi + 1];
                acc = __builtin_amdgcn_wmma_f32_16x16x4_f32(
                    false, af, false, bt[kc], (short)0, acc, false, false);
            }
            Cacc[t] = acc;
        }
        __builtin_amdgcn_wave_barrier();
#pragma unroll
        for (int t = 0; t < 4; ++t)
#pragma unroll
            for (int vg = 0; vg < 8; ++vg)
                Am[(16 * t + vg + 8 * hi) * PSTR + n] = Cacc[t][vg];
        __builtin_amdgcn_wave_barrier();
    }

    for (int i = 0; i < 32; ++i) {
        int d = i * 32 + lane;
        float v = Am[(d >> 4) * PSTR + (d & 15)];
        xc[d] = v; xb[d] = f2bf(v);
        __builtin_nontemporal_store(v, &xt[(size_t)d * (Ln + 1)]);
    }
}

// =====================================================================
// classify: Z = U @ S16 @ V^T (64x64) per sample via f32 WMMA,
// logits = Z . Wc + bc, softmax.  One wave per sample, 4 waves/block.
// =====================================================================
__global__ __launch_bounds__(128)
void classify_kernel(const float* __restrict__ Xcur,
                     const float* __restrict__ Wc,
                     const float* __restrict__ bc,
                     float* __restrict__ pred,
                     float* __restrict__ cls) {
    __shared__ float sU[4][DIMn * PSTR];   // holds U, then M = U@S16
    __shared__ float sS[4][16 * PSTR];
    __shared__ float sV[4][DIMn * PSTR];

    const int lane = threadIdx.x & 31;
    const int wave = threadIdx.x >> 5;
    const int b = blockIdx.x * 4 + wave;

    const float* Up = Xcur + (size_t)(b * 3 + 0) * Dn;
    const float* Sp = Xcur + (size_t)(b * 3 + 1) * Dn;
    const float* Vp = Xcur + (size_t)(b * 3 + 2) * Dn;
    float* U = sU[wave];
    float* S = sS[wave];
    float* V = sV[wave];

    __builtin_prefetch(Wc + lane * 64, 0, 1);           // warm Wc (global_prefetch_b8)

    for (int i = 0; i < 32; ++i) {
        int d = i * 32 + lane;
        U[(d >> 4) * PSTR + (d & 15)] = Up[d];
        V[(d >> 4) * PSTR + (d & 15)] = Vp[d];
    }
    for (int i = 0; i < 8; ++i) {
        int d = i * 32 + lane;
        S[(d >> 4) * PSTR + (d & 15)] = Sp[d];
    }
    __builtin_amdgcn_wave_barrier();

    const int n  = lane & 15;
    const int hi = lane >> 4;
    const v8f vz = {0.f,0.f,0.f,0.f,0.f,0.f,0.f,0.f};

    // ---- M = U @ S16 ----
    v2f bsf[4];
#pragma unroll
    for (int kc = 0; kc < 4; ++kc) {
        const int r0 = 4 * kc + 2 * hi;
        bsf[kc].x = S[r0 * PSTR + n];
        bsf[kc].y = S[(r0 + 1) * PSTR + n];
    }
    v8f Macc[4];
#pragma unroll
    for (int t = 0; t < 4; ++t) {
        v8f acc = vz;
#pragma unroll
        for (int kc = 0; kc < 4; ++kc) {
            v2f af;
            const int rr = 16 * t + n;
            af.x = U[rr * PSTR + 4 * kc + 2 * hi];
            af.y = U[rr * PSTR + 4 * kc + 2 * hi + 1];
            acc = __builtin_amdgcn_wmma_f32_16x16x4_f32(
                false, af, false, bsf[kc], (short)0, acc, false, false);
        }
        Macc[t] = acc;
    }
    __builtin_amdgcn_wave_barrier();
#pragma unroll
    for (int t = 0; t < 4; ++t)
#pragma unroll
        for (int vg = 0; vg < 8; ++vg)
            U[(16 * t + vg + 8 * hi) * PSTR + n] = Macc[t][vg];   // U := M
    __builtin_amdgcn_wave_barrier();

    // ---- Z = M @ V^T, fold into class dot products on the fly ----
    float part[NC];
#pragma unroll
    for (int c = 0; c < NC; ++c) part[c] = 0.0f;

    for (int t = 0; t < 4; ++t) {
        for (int ct = 0; ct < 4; ++ct) {
            v8f acc = vz;
#pragma unroll
            for (int kc = 0; kc < 4; ++kc) {
                v2f af, bf;
                const int rr = 16 * t + n;
                af.x = U[rr * PSTR + 4 * kc + 2 * hi];
                af.y = U[rr * PSTR + 4 * kc + 2 * hi + 1];
                const int vr = 16 * ct + n;             // B = V^T: B[k][n] = V[n][k]
                bf.x = V[vr * PSTR + 4 * kc + 2 * hi];
                bf.y = V[vr * PSTR + 4 * kc + 2 * hi + 1];
                acc = __builtin_amdgcn_wmma_f32_16x16x4_f32(
                    false, af, false, bf, (short)0, acc, false, false);
            }
#pragma unroll
            for (int vg = 0; vg < 8; ++vg) {
                const int zrow = 16 * t + vg + 8 * hi;
                const int zcol = 16 * ct + n;
                const int zi = zrow * DIMn + zcol;
                const float zv = acc[vg];
                for (int c = 0; c < NC; ++c)
                    part[c] += zv * Wc[c * (DIMn * DIMn) + zi];
            }
        }
    }
#pragma unroll
    for (int c = 0; c < NC; ++c)
        for (int off = 16; off; off >>= 1)
            part[c] += __shfl_xor(part[c], off, 32);

    float logits[NC], mx = -1e30f;
#pragma unroll
    for (int c = 0; c < NC; ++c) { logits[c] = part[c] + bc[c]; mx = fmaxf(mx, logits[c]); }
    float ex[NC], se = 0.0f;
#pragma unroll
    for (int c = 0; c < NC; ++c) { ex[c] = __expf(logits[c] - mx); se += ex[c]; }
    const float rse = 1.0f / se;
    if (lane == 0) {
#pragma unroll
        for (int c = 0; c < NC; ++c) {
            cls[(size_t)b * NC + c]  = logits[c];
            pred[(size_t)b * NC + c] = ex[c] * rse;
        }
    }
}

// =====================================================================
// launch
// =====================================================================
extern "C" void kernel_launch(void* const* d_in, const int* in_sizes, int n_in,
                              void* d_out, int out_size, void* d_ws, size_t ws_size,
                              hipStream_t stream) {
    const float* X  = (const float*)d_in[0];
    const float* h  = (const float*)d_in[1];
    const float* W0 = (const float*)d_in[2];
    const float* Ws = (const float*)d_in[3];
    const float* bs = (const float*)d_in[4];
    const float* Wc = (const float*)d_in[5];
    const float* bc = (const float*)d_in[6];

    // workspace layout (64 MB total)
    float* Xcur = (float*)d_ws;                          // 3072x1024 f32
    float* Y    = Xcur + (size_t)ROWS * Dn;              // 3072x1024 f32
    unsigned short* Xbf = (unsigned short*)(Y + (size_t)ROWS * Dn);   // bf16
    unsigned short* Wbf = Xbf + (size_t)ROWS * Dn;       // 17 x 1024x1024 bf16

    float* out  = (float*)d_out;
    float* pred = out;
    float* cls  = out + (size_t)Bn * NC;
    float* Xt   = out + (size_t)2 * Bn * NC;             // (B,3,D,L+1)

    prep_kernel<<<(ROWS * Dn) / 256, 256, 0, stream>>>(X, Xcur, Xbf);
    wconv_kernel<<<((Ln + 1) * Dn * Dn) / 256, 256, 0, stream>>>(W0, Ws, Wbf);

    // layer 0: plain linear, then projection
    gemm_kernel<<<dim3(Dn / 64, ROWS / 128), 128, 0, stream>>>(
        Xbf, Wbf, nullptr, nullptr, h, Y, 0);
    polar_kernel<<<ROWS / 8, 256, 0, stream>>>(Y, Xcur, Xbf, Xt, 0);

    // residual layers
    for (int l = 1; l <= Ln; ++l) {
        gemm_kernel<<<dim3(Dn / 64, ROWS / 128), 128, 0, stream>>>(
            Xbf, Wbf + (size_t)l * Dn * Dn, bs + (size_t)(l - 1) * Dn,
            Xcur, h, Y, 1);
        polar_kernel<<<ROWS / 8, 256, 0, stream>>>(Y, Xcur, Xbf, Xt, l);
    }

    classify_kernel<<<Bn / 4, 128, 0, stream>>>(Xcur, Wc, bc, pred, cls);
}
